// SparseMultiHeadAttention_27255862460459
// MI455X (gfx1250) — matrix operand order repaired and checked by reading, not compiled, not run on hardware
//
#include <hip/hip_runtime.h>

#define DEVI __device__ __forceinline__

typedef __bf16 bf16_t;
typedef bf16_t v16bf __attribute__((ext_vector_type(16)));
typedef bf16_t v8bf  __attribute__((ext_vector_type(8)));
typedef float  v8f   __attribute__((ext_vector_type(8)));

static constexpr int B_  = 4, S_ = 2048, D_ = 1024, H_ = 16, DK_ = 64;
static constexpr int TOPK = 204;              // max(1, int((1-0.9)*2048))
static constexpr int M_  = B_ * S_;           // 8192 rows
static constexpr int N3  = 3 * D_;            // 3072

// workspace byte offsets (total ~88 MB)
static constexpr size_t XB_OFF = 0;
static constexpr size_t WQ_OFF = XB_OFF + (size_t)M_ * D_ * 2;
static constexpr size_t WO_OFF = WQ_OFF + (size_t)N3 * D_ * 2;
static constexpr size_t Q_OFF  = WO_OFF + (size_t)D_ * D_ * 2;
static constexpr size_t K_OFF  = Q_OFF  + (size_t)B_ * H_ * S_ * DK_ * 2;
static constexpr size_t VT_OFF = K_OFF  + (size_t)B_ * H_ * S_ * DK_ * 2;
static constexpr size_t AO_OFF = VT_OFF + (size_t)B_ * H_ * S_ * DK_ * 2;

// attention kernel LDS: 16x2048 f32 scores + 16x256 hist + misc + 16x64 out acc
static constexpr int SC_B   = 16 * 2048 * 4;          // 131072
static constexpr int HIST_B = 16 * 256 * 4;           // 16384
static constexpr int MISC_B = 512;
static constexpr int OUT_B  = 16 * 64 * 4;            // 4096
static constexpr int ATTN_SMEM = SC_B + HIST_B + MISC_B + OUT_B; // 152064

// ---------------- WMMA helpers (bf16 in, f32 accumulate) ----------------

// A operand (16xK tile, row-major, ld in elements): lane<16 -> row=lane,
// K-chunks {k0..k0+7, k0+16..k0+23}; lane>=16 -> row=lane-16, chunks {+8,+24}.
DEVI v16bf load_a16(const bf16_t* base, int ld, int row0, int k0) {
  const int lane = threadIdx.x & 31;
  const int sel  = lane >> 4;
  const bf16_t* p = base + (size_t)(row0 + (lane & 15)) * ld + k0 + sel * 8;
  v8bf lo = *(const v8bf*)p;
  v8bf hi = *(const v8bf*)(p + 16);
  return __builtin_shufflevector(lo, hi, 0,1,2,3,4,5,6,7,8,9,10,11,12,13,14,15);
}

// B operand given as rows of B^T (N x K row-major): lane holds column n=lane&15,
// K-chunks {k0+(lane>>4)*8 .. +7, k0+16+(lane>>4)*8 .. +7}.
DEVI v16bf load_b16(const bf16_t* base, int ld, int row0, int k0) {
  const int lane = threadIdx.x & 31;
  const int sel  = lane >> 4;
  const bf16_t* p = base + (size_t)(row0 + (lane & 15)) * ld + k0 + sel * 8;
  v8bf lo = *(const v8bf*)p;
  v8bf hi = *(const v8bf*)(p + 16);
  return __builtin_shufflevector(lo, hi, 0,1,2,3,4,5,6,7,8,9,10,11,12,13,14,15);
}

// A operand gathered from f32 LDS (attention probabilities), converted to bf16.
DEVI v16bf load_a_lds(const float* base, int ld, int row0, int k0) {
  const int lane = threadIdx.x & 31;
  const int sel  = lane >> 4;
  const float* p = base + (row0 + (lane & 15)) * ld + k0 + sel * 8;
  v16bf r;
#pragma unroll
  for (int i = 0; i < 8; ++i) {
    r[i]     = (bf16_t)p[i];
    r[i + 8] = (bf16_t)p[i + 16];
  }
  return r;
}

DEVI v8f wmma_bf16(v16bf a, v16bf b, v8f c) {
  return __builtin_amdgcn_wmma_f32_16x16x32_bf16(false, a, false, b, (short)0, c,
                                                 false, false);
}

// ---------------- kernels ----------------

__global__ void cvt_kernel(const float* __restrict__ in, bf16_t* __restrict__ out, int n) {
  for (int i = blockIdx.x * blockDim.x + threadIdx.x; i < n; i += gridDim.x * blockDim.x)
    out[i] = (bf16_t)in[i];
}

// qkv = x @ Wqkv^T + bqkv ; scatter into head-major Q, K and transposed V.
__global__ __launch_bounds__(256) void qkv_gemm(const bf16_t* __restrict__ Xb,
                                                const bf16_t* __restrict__ Wq,
                                                const float* __restrict__ bias,
                                                bf16_t* __restrict__ Qm,
                                                bf16_t* __restrict__ Km,
                                                bf16_t* __restrict__ VTm) {
  const int w = threadIdx.x >> 5, lane = threadIdx.x & 31;
  const int m0 = blockIdx.x * 16;
  const int n0 = (blockIdx.y * 8 + w) * 64;
  const v8f z = {0.f,0.f,0.f,0.f,0.f,0.f,0.f,0.f};
  v8f c[4] = {z, z, z, z};
  for (int kk = 0; kk < D_; kk += 32) {
    v16bf a = load_a16(Xb, D_, m0, kk);
#pragma unroll
    for (int j = 0; j < 4; ++j)
      c[j] = wmma_bf16(a, load_b16(Wq, D_, n0 + j * 16, kk), c[j]);
  }
  const int nlo = lane & 15, mrow = (lane >> 4) * 8;
#pragma unroll
  for (int j = 0; j < 4; ++j) {
    const int n = n0 + j * 16 + nlo;
    const int which = n >> 10, dg = n & 1023;
    const int hh = dg >> 6, dk = dg & 63;
    const float bv = bias[n];
#pragma unroll
    for (int r = 0; r < 8; ++r) {
      const int m = m0 + mrow + r;
      const int bb = m >> 11, s = m & 2047;
      const size_t bh = (size_t)(bb * H_ + hh);
      const bf16_t val = (bf16_t)(c[j][r] + bv);
      if (which == 0)      Qm[(bh * S_ + s) * DK_ + dk] = val;
      else if (which == 1) Km[(bh * S_ + s) * DK_ + dk] = val;
      else                 VTm[(bh * DK_ + dk) * S_ + s] = val;
    }
  }
}

// One workgroup per (b, h, 16-query tile): scores -> exact top-k threshold
// (radix select) -> masked softmax -> attn @ V, all resident in LDS.
__global__ __launch_bounds__(256) void attn_kernel(const bf16_t* __restrict__ Qm,
                                                   const bf16_t* __restrict__ Km,
                                                   const bf16_t* __restrict__ VTm,
                                                   bf16_t* __restrict__ AO) {
  extern __shared__ char smem[];
  float*    sc     = (float*)smem;                           // [16][2048]
  unsigned* hist   = (unsigned*)(smem + SC_B);               // [16][256]
  unsigned* prefix = (unsigned*)(smem + SC_B + HIST_B);      // [16]
  int*      kcur   = (int*)(smem + SC_B + HIST_B + 64);      // [16]
  float*    thr    = (float*)(smem + SC_B + HIST_B + 128);   // [16]
  float*    outacc = (float*)(smem + SC_B + HIST_B + MISC_B);// [16*64]

  const int tid = threadIdx.x;
  const int w = tid >> 5, lane = tid & 31;
  const int qt = blockIdx.x & 127;   // S_/16 tiles
  const int bh = blockIdx.x >> 7;    // 0..63
  const int b = bh >> 4, h = bh & 15;
  const int q0 = qt * 16;

  const bf16_t* Qb = Qm  + (size_t)bh * S_ * DK_;
  const bf16_t* Kb = Km  + (size_t)bh * S_ * DK_;
  const bf16_t* Vb = VTm + (size_t)bh * DK_ * S_;

  // ---- phase 1: scores tile 16 x 2048, each wave owns a 256-wide strip ----
  {
    v16bf a0 = load_a16(Qb, DK_, q0, 0);
    v16bf a1 = load_a16(Qb, DK_, q0, 32);
    const int s0 = w * 256;
    const int scol = lane & 15, mrow = (lane >> 4) * 8;
    for (int nt = 0; nt < 16; ++nt) {
      const int srow = s0 + nt * 16;
      v8f c = {0.f,0.f,0.f,0.f,0.f,0.f,0.f,0.f};
      c = wmma_bf16(a0, load_b16(Kb, DK_, srow, 0),  c);
      c = wmma_bf16(a1, load_b16(Kb, DK_, srow, 32), c);
#pragma unroll
      for (int r = 0; r < 8; ++r)
        sc[(mrow + r) * S_ + srow + scol] = c[r] * 0.125f;  // 1/sqrt(64)
    }
  }
  if (tid < 16) { prefix[tid] = 0u; kcur[tid] = TOPK; }
  __syncthreads();

  // ---- phase 2: 4-pass radix select of the 204th-largest value per row ----
  for (int pass = 0; pass < 4; ++pass) {
    const int shift = 24 - 8 * pass;
    for (int i = tid; i < 16 * 256; i += 256) hist[i] = 0u;
    __syncthreads();
    const unsigned hmask = (pass == 0) ? 0u : (0xFFFFFFFFu << (shift + 8));
    for (int rr = 0; rr < 2; ++rr) {
      const int r = w * 2 + rr;                 // wave handles rows 2w, 2w+1
      const unsigned pfx = prefix[r];
      for (int i = 0; i < 64; ++i) {
        unsigned u = __float_as_uint(sc[r * S_ + lane + i * 32]);
        u = (u & 0x80000000u) ? ~u : (u | 0x80000000u);   // order-preserving key
        if ((u & hmask) == pfx)
          atomicAdd(&hist[r * 256 + ((u >> shift) & 255u)], 1u);
      }
    }
    __syncthreads();
    if (tid < 16) {
      const int r = tid;
      int need = kcur[r];
      unsigned c = 0; int sel = 0;
      for (int bin = 255; bin >= 0; --bin) {
        const unsigned hc = hist[r * 256 + bin];
        if (c + hc >= (unsigned)need) { sel = bin; need -= (int)c; break; }
        c += hc;
      }
      kcur[r] = need;
      prefix[r] |= (unsigned)sel << shift;
    }
    __syncthreads();
  }
  if (tid < 16) {
    const unsigned key = prefix[tid];
    const unsigned bits = (key & 0x80000000u) ? (key & 0x7FFFFFFFu) : ~key;
    thr[tid] = __uint_as_float(bits);
  }
  __syncthreads();

  // ---- phase 3: softmax over (score>=thr ? score : 0), wave32 reductions ----
  for (int rr = 0; rr < 2; ++rr) {
    const int r = w * 2 + rr;
    const float t = thr[r];
    float mx = -3.0e38f;
    for (int i = 0; i < 64; ++i) {
      float v = sc[r * S_ + lane + i * 32];
      v = (v >= t) ? v : 0.f;
      sc[r * S_ + lane + i * 32] = v;
      mx = fmaxf(mx, v);
    }
    for (int off = 16; off > 0; off >>= 1)
      mx = fmaxf(mx, __shfl_xor(mx, off, 32));
    float sum = 0.f;
    for (int i = 0; i < 64; ++i) {
      const float e = __expf(sc[r * S_ + lane + i * 32] - mx);
      sc[r * S_ + lane + i * 32] = e;
      sum += e;
    }
    for (int off = 16; off > 0; off >>= 1)
      sum += __shfl_xor(sum, off, 32);
    const float inv = 1.f / (sum + 1e-9f);
    for (int i = 0; i < 64; ++i)
      sc[r * S_ + lane + i * 32] *= inv;
  }
  for (int i = tid; i < 16 * 64; i += 256) outacc[i] = 0.f;
  __syncthreads();

  // ---- phase 4: out(16x64) = attn(16x2048) @ V, wave-split over K ----
  {
    const v8f z = {0.f,0.f,0.f,0.f,0.f,0.f,0.f,0.f};
    v8f acc[4] = {z, z, z, z};
    const int s0 = w * 256;
    for (int kk = 0; kk < 256; kk += 32) {
      v16bf a = load_a_lds(sc, S_, 0, s0 + kk);
#pragma unroll
      for (int nt = 0; nt < 4; ++nt)
        acc[nt] = wmma_bf16(a, load_b16(Vb, S_, nt * 16, s0 + kk), acc[nt]);
    }
    const int nlo = lane & 15, mrow = (lane >> 4) * 8;
#pragma unroll
    for (int nt = 0; nt < 4; ++nt)
#pragma unroll
      for (int r = 0; r < 8; ++r)
        atomicAdd(&outacc[(mrow + r) * 64 + nt * 16 + nlo], acc[nt][r]);
  }
  __syncthreads();

  // ---- phase 5: write bf16 attention output in (B,S,D) layout ----
  for (int i = tid; i < 16 * 64; i += 256) {
    const int m = i >> 6, d = i & 63;
    AO[((size_t)(b * S_ + q0 + m)) * D_ + h * DK_ + d] = (bf16_t)outacc[i];
  }
}

// final: out = attnOut @ Wo^T + bo (f32 result)
__global__ __launch_bounds__(256) void out_gemm(const bf16_t* __restrict__ A,
                                                const bf16_t* __restrict__ Wb,
                                                const float* __restrict__ bias,
                                                float* __restrict__ out) {
  const int w = threadIdx.x >> 5, lane = threadIdx.x & 31;
  const int m0 = blockIdx.x * 16;
  const int n0 = (blockIdx.y * 8 + w) * 64;
  const v8f z = {0.f,0.f,0.f,0.f,0.f,0.f,0.f,0.f};
  v8f c[4] = {z, z, z, z};
  for (int kk = 0; kk < D_; kk += 32) {
    v16bf a = load_a16(A, D_, m0, kk);
#pragma unroll
    for (int j = 0; j < 4; ++j)
      c[j] = wmma_bf16(a, load_b16(Wb, D_, n0 + j * 16, kk), c[j]);
  }
  const int nlo = lane & 15, mrow = (lane >> 4) * 8;
#pragma unroll
  for (int j = 0; j < 4; ++j) {
    const int n = n0 + j * 16 + nlo;
    const float bv = bias[n];
#pragma unroll
    for (int r = 0; r < 8; ++r)
      out[(size_t)(m0 + mrow + r) * D_ + n] = c[j][r] + bv;
  }
}

extern "C" void kernel_launch(void* const* d_in, const int* in_sizes, int n_in,
                              void* d_out, int out_size, void* d_ws, size_t ws_size,
                              hipStream_t stream) {
  (void)in_sizes; (void)n_in; (void)out_size; (void)ws_size;
  const float* x    = (const float*)d_in[0];
  const float* Wqkv = (const float*)d_in[1];
  const float* bqkv = (const float*)d_in[2];
  const float* Wo   = (const float*)d_in[3];
  const float* bo   = (const float*)d_in[4];

  char* ws = (char*)d_ws;
  bf16_t* Xb  = (bf16_t*)(ws + XB_OFF);
  bf16_t* Wqb = (bf16_t*)(ws + WQ_OFF);
  bf16_t* Wob = (bf16_t*)(ws + WO_OFF);
  bf16_t* Qm  = (bf16_t*)(ws + Q_OFF);
  bf16_t* Km  = (bf16_t*)(ws + K_OFF);
  bf16_t* VTm = (bf16_t*)(ws + VT_OFF);
  bf16_t* AO  = (bf16_t*)(ws + AO_OFF);

  cvt_kernel<<<2048, 256, 0, stream>>>(x,    Xb,  M_ * D_);
  cvt_kernel<<<2048, 256, 0, stream>>>(Wqkv, Wqb, N3 * D_);
  cvt_kernel<<<1024, 256, 0, stream>>>(Wo,   Wob, D_ * D_);

  qkv_gemm<<<dim3(M_ / 16, N3 / 512), 256, 0, stream>>>(Xb, Wqb, bqkv, Qm, Km, VTm);

  hipFuncSetAttribute(reinterpret_cast<const void*>(attn_kernel),
                      hipFuncAttributeMaxDynamicSharedMemorySize, ATTN_SMEM);
  attn_kernel<<<B_ * H_ * (S_ / 16), 256, ATTN_SMEM, stream>>>(Qm, Km, VTm, AO);

  out_gemm<<<dim3(M_ / 16, D_ / 512), 256, 0, stream>>>(AO, Wob, bo, (float*)d_out);
}
